// TFNFullLayer_2302102471545
// MI455X (gfx1250) — matrix operand order repaired and checked by reading, not compiled, not run on hardware
//
#include <hip/hip_runtime.h>
#include <cstdint>
#include <cstddef>

// ---------------- types / constants ----------------
typedef _Float16 v16h __attribute__((ext_vector_type(16)));
typedef float    v8f  __attribute__((ext_vector_type(8)));

#define IN_DIM   80
#define MUL0     32
#define MUL1     16
#define HID      128
#define NUM_RBF  24
#define WN       2304       // W_NUMEL
#define NCT      144        // WN / 16 column tiles
#define NGRP     18         // 144 / 8 tiles per staged group

#define C_ALPHA     0.14433756729740643f   // 1/sqrt(48)
#define C_INVSQRT3  0.57735026918962584f
#define C_RS24      0.20412414523193154f   // 1/sqrt(24)
#define C_RS128     0.08838834764831845f   // 1/sqrt(128)
#define C_SQRT3     1.7320508075688772f

union H16 { uint4 q[2]; v16h v; };

__device__ __forceinline__ float silu_f(float z) { return z / (1.0f + __expf(-z)); }

// ---------------- kernel: zero accumulator ----------------
__global__ void k_zero_agg(float* __restrict__ agg, int n) {
  int i = blockIdx.x * blockDim.x + threadIdx.x;
  if (i < n) agg[i] = 0.0f;
}

// ---------------- kernel: pre-swizzle W3 into WMMA B-fragment layout (f16) ----------------
// B frag (16-bit, 32x16), k order chosen here (same as A): lane L holds column n=L%16, K = (L/16)*16 + i  (i=0..15 contiguous)
// swizzled offset o = (((ct*4 + kc)*32 + lane)*16 + i)  ->  fragment-linear, group-contiguous
__global__ void k_prep_w3(const float* __restrict__ W3, _Float16* __restrict__ W3s) {
  int o = blockIdx.x * blockDim.x + threadIdx.x;
  if (o >= NCT * 4 * 32 * 16) return;
  int i    = o & 15;
  int lane = (o >> 4) & 31;
  int kc   = (o >> 9) & 3;
  int ct   = o >> 11;
  int k    = kc * 32 + (lane >> 4) * 16 + i;
  int col  = ct * 16 + (lane & 15);
  W3s[o] = (_Float16)(W3[(size_t)k * WN + col] * C_RS128);  // fold 1/sqrt(HID)
}

// ---------------- kernel: per-edge RBF + 2-layer MLP -> h2 (f16) ----------------
__global__ __launch_bounds__(64) void k_edge_mlp(
    const float* __restrict__ pos, const int* __restrict__ ei, int E,
    const float* __restrict__ W1, const float* __restrict__ W2,
    _Float16* __restrict__ h2out) {
  __shared__ float h1s[64][132];   // padded: bank-friendly, 16B-aligned rows
  const int tid = threadIdx.x;
  int e = blockIdx.x * 64 + tid;
  const bool valid = e < E;
  int ec = valid ? e : (E - 1);
  int r0 = ei[ec], c0 = ei[E + ec];
  float dx = pos[r0 * 3 + 0] - pos[c0 * 3 + 0];
  float dy = pos[r0 * 3 + 1] - pos[c0 * 3 + 1];
  float dz = pos[r0 * 3 + 2] - pos[c0 * 3 + 2];
  float r = sqrtf(dx * dx + dy * dy + dz * dz + 1e-12f);

  float phi[NUM_RBF];
  const float sig_inv = (float)NUM_RBF / 10.0f;          // 1/sigma
  #pragma unroll
  for (int j = 0; j < NUM_RBF; ++j) {
    float mu = (10.0f / 23.0f) * (float)j;               // linspace(0,10,24)
    float t = (r - mu) * sig_inv;
    phi[j] = __expf(-t * t);
  }
  // layer 1: h1 = silu(phi @ W1 / sqrt(24)); W1 indices wave-uniform -> scalar loads
  for (int k = 0; k < HID; ++k) {
    float acc = 0.0f;
    #pragma unroll
    for (int j = 0; j < NUM_RBF; ++j) acc += phi[j] * W1[j * HID + k];
    h1s[tid][k] = silu_f(acc * C_RS24);
  }
  // layer 2: h2 = silu(h1 @ W2 / sqrt(128))
  for (int kk = 0; kk < HID; ++kk) {
    float acc = 0.0f;
    for (int k = 0; k < HID; k += 4) {
      float4 h4 = *(const float4*)&h1s[tid][k];
      acc += h4.x * W2[(k + 0) * HID + kk];
      acc += h4.y * W2[(k + 1) * HID + kk];
      acc += h4.z * W2[(k + 2) * HID + kk];
      acc += h4.w * W2[(k + 3) * HID + kk];
    }
    if (valid) h2out[(size_t)e * HID + kk] = (_Float16)silu_f(acc * C_RS128);
  }
}

// ---------------- kernel: fused WMMA GEMM (h2 @ W3) + equivariant contraction + scatter ----------------
struct WaveLds {
  float sA[16][MUL0];       // raw scalar features s[m][u]
  float vA[16][MUL1][3];    // raw vector features v[m][u][xyz]
  float bC[16][MUL1];       // INV_SQRT3 * (v[m][u] . Y1[m])
  float y1[16][3];
  int   rowi[16];
};

// async linear copy: one 32KB group (8 column tiles) of swizzled W3 -> LDS.
// 256 threads x 128B each; INST_OFFSET applies to both LDS and global address.
__device__ __forceinline__ void stage_group(const _Float16* __restrict__ W3s,
                                            uint32_t ldsDst, int g, int tid) {
  uint64_t src = (uint64_t)(uintptr_t)(W3s + (size_t)g * 16384) + (uint32_t)tid * 128u;
  uint32_t dst = ldsDst + (uint32_t)tid * 128u;
  asm volatile(
      "global_load_async_to_lds_b128 %0, %1, off offset:0\n\t"
      "global_load_async_to_lds_b128 %0, %1, off offset:16\n\t"
      "global_load_async_to_lds_b128 %0, %1, off offset:32\n\t"
      "global_load_async_to_lds_b128 %0, %1, off offset:48\n\t"
      "global_load_async_to_lds_b128 %0, %1, off offset:64\n\t"
      "global_load_async_to_lds_b128 %0, %1, off offset:80\n\t"
      "global_load_async_to_lds_b128 %0, %1, off offset:96\n\t"
      "global_load_async_to_lds_b128 %0, %1, off offset:112"
      :: "v"(dst), "v"(src) : "memory");
}

__device__ __forceinline__ void wait_async0() {
  asm volatile("s_wait_asynccnt 0x0" ::: "memory");
}

__global__ __launch_bounds__(256) void k_gemm_contract(
    const float* __restrict__ x, const float* __restrict__ pos,
    const int* __restrict__ ei, int E,
    const _Float16* __restrict__ h2, const _Float16* __restrict__ W3s,
    float* __restrict__ agg) {
  __shared__ WaveLds lds[8];
  __shared__ alignas(16) _Float16 Bstage[2][16384];   // 2 x 32KB double buffer
  const int tid  = threadIdx.x;
  const int wv   = tid >> 5;
  const int lane = tid & 31;
  WaveLds& L = lds[wv];
  const int baseE = (blockIdx.x * 8 + wv) * 16;

  // ---- setup: edge geometry ----
  if (lane < 16) {
    int e = baseE + lane; if (e >= E) e = E - 1;
    int r0 = ei[e], c0 = ei[E + e];
    L.rowi[lane] = r0;
    float dx = pos[r0 * 3 + 0] - pos[c0 * 3 + 0];
    float dy = pos[r0 * 3 + 1] - pos[c0 * 3 + 1];
    float dz = pos[r0 * 3 + 2] - pos[c0 * 3 + 2];
    float rr = sqrtf(dx * dx + dy * dy + dz * dz + 1e-12f);
    float sc = C_SQRT3 / rr;
    L.y1[lane][0] = dx * sc; L.y1[lane][1] = dy * sc; L.y1[lane][2] = dz * sc;
  }
  __syncthreads();
  // ---- gather x[row] for the 16 edges ----
  for (int i = lane; i < 16 * IN_DIM; i += 32) {
    int m = i / IN_DIM, q = i - m * IN_DIM;
    float val = x[(size_t)L.rowi[m] * IN_DIM + q];
    if (q < MUL0) L.sA[m][q] = val;
    else { int t = q - MUL0; L.vA[m][t / 3][t % 3] = val; }
  }
  __syncthreads();
  // ---- precompute vd coefficients ----
  for (int i = lane; i < 16 * MUL1; i += 32) {
    int m = i >> 4, u = i & 15;
    L.bC[m][u] = C_INVSQRT3 * (L.vA[m][u][0] * L.y1[m][0] +
                               L.vA[m][u][1] * L.y1[m][1] +
                               L.vA[m][u][2] * L.y1[m][2]);
  }
  __syncthreads();

  // ---- A fragments: 16x128 f16, registers for all 144 column tiles ----
  // A in this code's contiguous k order: lane<16 holds M=lane, K 0-15; lane>=16 holds K 16-31 (matches B swizzle)
  const int mrow = lane & 15;
  const int half = lane >> 4;
  const int srow = half * 8;
  H16 af[4];
  {
    int ea = baseE + mrow; if (ea >= E) ea = E - 1;
    const _Float16* ap = h2 + (size_t)ea * HID;
    #pragma unroll
    for (int kc = 0; kc < 4; ++kc) {
      const _Float16* p = ap + kc * 32 + half * 16;
      af[kc].q[0] = *(const uint4*)(p);
      af[kc].q[1] = *(const uint4*)(p + 8);
    }
  }
  float y1r[8][3];
  #pragma unroll
  for (int r = 0; r < 8; ++r)
    #pragma unroll
    for (int mm = 0; mm < 3; ++mm) y1r[r][mm] = L.y1[srow + r][mm];

  float accLo[8] = {};      // out0 cols 0..15   (col = n)
  float accHi[8] = {};      // out0 cols 16..31  (col = 16+n)
  float acc1[8][3] = {};    // out1[w_=n][xyz]

  const uint32_t bufOff[2] = { (uint32_t)(uintptr_t)&Bstage[0][0],
                               (uint32_t)(uintptr_t)&Bstage[1][0] };

  // one 16x16 w-tile from LDS buffer: 4 chained K=32 WMMAs
  auto tileL = [&](int buf, int lct) -> v8f {
    v8f c = {0.f, 0.f, 0.f, 0.f, 0.f, 0.f, 0.f, 0.f};
    #pragma unroll
    for (int kc = 0; kc < 4; ++kc) {
      const _Float16* bp = &Bstage[buf][(((lct * 4 + kc) * 32 + lane) << 4)];
      H16 b;
      b.q[0] = *(const uint4*)(bp);
      b.q[1] = *(const uint4*)(bp + 8);
      c = __builtin_amdgcn_wmma_f32_16x16x32_f16(
              false, af[kc].v, false, b.v, (short)0, c, false, false);
    }
    return c;
  };

  // prologue: stage group 0
  stage_group(W3s, bufOff[0], 0, tid);

  for (int g = 0; g < NGRP; ++g) {
    wait_async0();          // our share of buffer g is in LDS
    __syncthreads();        // everyone's share is in LDS; prior buffer fully consumed
    if (g + 1 < NGRP) stage_group(W3s, bufOff[(g + 1) & 1], g + 1, tid);
    const int buf = g & 1;

    if (g < 8) {
      // w_ss: ct = 8g+lct, u = 4g + lct/2, coeff = s[m][u]
      #pragma unroll
      for (int lp = 0; lp < 4; ++lp) {
        int u = 4 * g + lp;
        v8f c0 = tileL(buf, 2 * lp);
        #pragma unroll
        for (int r = 0; r < 8; ++r) accLo[r] += L.sA[srow + r][u] * c0[r];
        v8f c1 = tileL(buf, 2 * lp + 1);
        #pragma unroll
        for (int r = 0; r < 8; ++r) accHi[r] += L.sA[srow + r][u] * c1[r];
      }
    } else if (g < 12) {
      // w_vd: coeff = INV_SQRT3*(v.Y1)[m][u], u = 4(g-8) + lct/2
      #pragma unroll
      for (int lp = 0; lp < 4; ++lp) {
        int u = 4 * (g - 8) + lp;
        v8f c0 = tileL(buf, 2 * lp);
        #pragma unroll
        for (int r = 0; r < 8; ++r) accLo[r] += L.bC[srow + r][u] * c0[r];
        v8f c1 = tileL(buf, 2 * lp + 1);
        #pragma unroll
        for (int r = 0; r < 8; ++r) accHi[r] += L.bC[srow + r][u] * c1[r];
      }
    } else if (g < 16) {
      // w_sv: u = 8(g-12)+lct; out1 += s[m][u]*Y1[m][mm]*w
      #pragma unroll
      for (int lct = 0; lct < 8; ++lct) {
        int u = 8 * (g - 12) + lct;
        v8f c = tileL(buf, lct);
        #pragma unroll
        for (int r = 0; r < 8; ++r) {
          float t0 = L.sA[srow + r][u] * c[r];
          acc1[r][0] += y1r[r][0] * t0;
          acc1[r][1] += y1r[r][1] * t0;
          acc1[r][2] += y1r[r][2] * t0;
        }
      }
    } else {
      // w_vs: u = 8(g-16)+lct; out1 += v[m][u][mm]*w
      #pragma unroll
      for (int lct = 0; lct < 8; ++lct) {
        int u = 8 * (g - 16) + lct;
        v8f c = tileL(buf, lct);
        #pragma unroll
        for (int r = 0; r < 8; ++r) {
          acc1[r][0] += L.vA[srow + r][u][0] * c[r];
          acc1[r][1] += L.vA[srow + r][u][1] * c[r];
          acc1[r][2] += L.vA[srow + r][u][2] * c[r];
        }
      }
    }
  }

  // ---- scatter-add messages into agg[row] ----
  const int n = lane & 15;
  #pragma unroll
  for (int r = 0; r < 8; ++r) {
    int m = srow + r;
    int e = baseE + m;
    if (e < E) {
      float* dst = agg + (size_t)L.rowi[m] * IN_DIM;
      unsafeAtomicAdd(dst + n,        C_ALPHA * accLo[r]);
      unsafeAtomicAdd(dst + 16 + n,   C_ALPHA * accHi[r]);
      #pragma unroll
      for (int mm = 0; mm < 3; ++mm)
        unsafeAtomicAdd(dst + MUL0 + 3 * n + mm, (C_ALPHA * C_INVSQRT3) * acc1[r][mm]);
    }
  }
}

// ---------------- kernel: final SiLU on scalar channel ----------------
__global__ void k_finish(const float* __restrict__ agg, float* __restrict__ out, int total) {
  int i = blockIdx.x * blockDim.x + threadIdx.x;
  if (i >= total) return;
  float v = agg[i];
  out[i] = ((i % IN_DIM) < MUL0) ? silu_f(v) : v;
}

// ---------------- launcher ----------------
extern "C" void kernel_launch(void* const* d_in, const int* in_sizes, int n_in,
                              void* d_out, int out_size, void* d_ws, size_t ws_size,
                              hipStream_t stream) {
  const float* x   = (const float*)d_in[0];
  const float* pos = (const float*)d_in[1];
  const int*   ei  = (const int*)d_in[2];
  const float* W1  = (const float*)d_in[3];
  const float* W2  = (const float*)d_in[4];
  const float* W3  = (const float*)d_in[5];
  const int N = in_sizes[0] / IN_DIM;
  const int E = in_sizes[2] / 2;
  float* out = (float*)d_out;

  char* ws = (char*)d_ws;
  size_t off = 0;
  float* agg = (float*)(ws + off);
  off += (((size_t)N * IN_DIM * sizeof(float)) + 255) & ~(size_t)255;
  _Float16* W3s = (_Float16*)(ws + off);
  off += (((size_t)NCT * 4 * 32 * 16 * sizeof(_Float16)) + 255) & ~(size_t)255;
  _Float16* h2 = (_Float16*)(ws + off);
  off += (((size_t)E * HID * sizeof(_Float16)) + 255) & ~(size_t)255;
  (void)ws_size; (void)n_in; (void)out_size;

  const int totalAgg = N * IN_DIM;
  k_zero_agg<<<(totalAgg + 255) / 256, 256, 0, stream>>>(agg, totalAgg);
  const int w3n = NCT * 4 * 32 * 16;
  k_prep_w3<<<(w3n + 255) / 256, 256, 0, stream>>>(W3, W3s);
  k_edge_mlp<<<(E + 63) / 64, 64, 0, stream>>>(pos, ei, E, W1, W2, h2);
  const int nBlocks2 = (E + 127) / 128;   // 8 waves/block, 16 edges/wave
  k_gemm_contract<<<nBlocks2, 256, 0, stream>>>(x, pos, ei, E, h2, W3s, agg);
  k_finish<<<(totalAgg + 255) / 256, 256, 0, stream>>>(agg, out, totalAgg);
}
